// Experts_27333171871856
// MI455X (gfx1250) — matrix operand order repaired and checked by reading, not compiled, not run on hardware
//
#include <hip/hip_runtime.h>

// ---------------------------------------------------------------------------
// Fused MoE SwiGLU for MI455X (gfx1250, wave32, WMMA bf16).
//   out[b,s,k,h] = (silu(x@Wg[k]) * (x@Wu[k])) @ Wd[k]
// B=4 S=2048 H=4096 E=1792 K=4 -> M = 8192 token rows, ~1.44 TFLOP.
// One workgroup = one expert x one 64-row tile. act[64][E] (bf16) stays in
// LDS. Global f32 tiles are register-pipelined (fetch -> compute -> convert),
// staged in LDS as bf16, consumed as ds_load_b128 fragments. Each wave owns a
// 2x2 block of 16x16 tiles so every fragment feeds two WMMAs. Global loads
// use uniform SGPR bases + loop-invariant per-thread offsets (GVS mode).
// ---------------------------------------------------------------------------

#define BDIM  4
#define SDIM  2048
#define HDIM  4096
#define EDIM  1792
#define KSEL  4
#define MROWS (BDIM * SDIM)      // 8192

#define MT    64                 // rows per block
#define XLD   72                 // x tile stride   (elems): 16B rows, gcd(36,64)=4
#define WLD   88                 // w tile stride   (elems): 16B rows, gcd(44,64)=4
#define ALD   1800               // act tile stride (elems): 16B rows, gcd(900,64)=4

typedef __attribute__((ext_vector_type(16))) __bf16 v16bf;
typedef __attribute__((ext_vector_type(8)))  float  v8f;

union Frag16 { uint4 q[2]; v16bf v; };

__device__ __forceinline__ unsigned short f2bf(float f) {
  __bf16 h = (__bf16)f;                       // RNE f32->bf16
  return __builtin_bit_cast(unsigned short, h);
}
__device__ __forceinline__ unsigned int pk2bf(float a, float b) {
  return (unsigned int)f2bf(a) | ((unsigned int)f2bf(b) << 16);
}
// two bf16 lane-stores from one packed dword (b16 + b16_d16_hi)
__device__ __forceinline__ void st2(unsigned short* p0, unsigned short* p1,
                                    unsigned int v) {
  p0[0] = (unsigned short)v;
  p1[0] = (unsigned short)(v >> 16);
}
__device__ __forceinline__ float4 ldg4(const float* base, int byteOff) {
  return *(const float4*)((const char*)base + byteOff);   // saddr + voffset
}

// A fragment (16x32 bf16 MxK): one contiguous 32B run per lane.
__device__ __forceinline__ v16bf frag_a(const uint4* base, int q0) {
  Frag16 f; f.q[0] = base[q0]; f.q[1] = base[q0 + 1]; return f.v;
}
// B fragment (32x16 bf16 KxN) from K-contiguous ([n][k]) tile: one 32B run.
__device__ __forceinline__ v16bf frag_b(const uint4* base, int q0) {
  Frag16 f; f.q[0] = base[q0]; f.q[1] = base[q0 + 1]; return f.v;
}

__device__ __forceinline__ v8f wmma_bf16(v16bf a, v16bf b, v8f c) {
  return __builtin_amdgcn_wmma_f32_16x16x32_bf16(false, a, false, b, (short)0,
                                                 c, false, false);
}

// -------- register-pipelined tiles: raw f32 (fetch) and packed bf16 (store) --
struct Raw1 { float4 xv[4]; float4 gv[8]; float4 uv[8]; };
struct Pk1  { uint2  xv[4]; uint2  gv[8]; uint2  uv[8]; };
struct Raw2 { float4 dv[8]; };
struct Pk2  { uint2  dv[8]; };

__device__ __forceinline__ void cvt1(Pk1& p, const Raw1& r) {
#pragma unroll
  for (int i = 0; i < 4; ++i)
    p.xv[i] = make_uint2(pk2bf(r.xv[i].x, r.xv[i].y), pk2bf(r.xv[i].z, r.xv[i].w));
#pragma unroll
  for (int i = 0; i < 8; ++i) {
    p.gv[i] = make_uint2(pk2bf(r.gv[i].x, r.gv[i].y), pk2bf(r.gv[i].z, r.gv[i].w));
    p.uv[i] = make_uint2(pk2bf(r.uv[i].x, r.uv[i].y), pk2bf(r.uv[i].z, r.uv[i].w));
  }
}
__device__ __forceinline__ void cvt2(Pk2& p, const Raw2& r) {
#pragma unroll
  for (int i = 0; i < 8; ++i)
    p.dv[i] = make_uint2(pk2bf(r.dv[i].x, r.dv[i].y), pk2bf(r.dv[i].z, r.dv[i].w));
}

__global__ void __launch_bounds__(256)
moe_swiglu_fused(const float* __restrict__ x,
                 const float* __restrict__ gate,
                 const float* __restrict__ up,
                 const float* __restrict__ down,
                 const int*   __restrict__ eidx,
                 float*       __restrict__ out) {
  extern __shared__ unsigned short smem[];
  unsigned short* actS = smem;                       // 64*ALD
  unsigned short* xS   = actS + MT * ALD;            // 64*XLD
  unsigned short* wgT  = xS + MT * XLD;              // 128*WLD
  unsigned short* wuT  = wgT + 128 * WLD;            // 128*WLD
  unsigned short* wdT  = wgT;                        // phase-2 reuses gate slot

  const int m0 = blockIdx.x * MT;
  const int e  = blockIdx.y;
  const int ke = eidx[e];
  const float* gW = gate + (size_t)ke * HDIM * EDIM;
  const float* uW = up   + (size_t)ke * HDIM * EDIM;
  const float* dW = down + (size_t)ke * EDIM * HDIM;

  const int t      = threadIdx.x;
  const int w      = t >> 5;             // wave 0..7
  const int lane   = t & 31;
  const int half   = lane >> 4;
  const int lane16 = lane & 15;
  // 2x2 block of 16x16 tiles per wave
  const int mt0 = (w & 1) * 32;          // m tiles {mt0, mt0+16}
  const int nb0 = (w >> 1) * 32;         // n tiles {nb0, nb0+16}

  // Loop-invariant per-thread global byte offsets (GVS: saddr + voffset)
  int offX[4], offW[8], offD[8];
#pragma unroll
  for (int i = 0; i < 4; ++i) {
    int q = t + i * 256;
    offX[i] = (((q >> 4) * HDIM) + ((q & 15) << 2)) * 4;
  }
#pragma unroll
  for (int i = 0; i < 8; ++i) {
    int q = t + i * 256, row = q >> 5, col = (q & 31) << 2;
    offW[i] = (row * EDIM + col) * 4;
    offD[i] = (row * HDIM + col) * 4;
  }
  // Loop-invariant LDS store addresses
  unsigned short* xSt[4];
  unsigned short* wRow[8];               // &tile[col*WLD + row], tile-relative
  int wOff[8];
#pragma unroll
  for (int i = 0; i < 4; ++i) {
    int q = t + i * 256;
    xSt[i] = xS + (q >> 4) * XLD + ((q & 15) << 2);
  }
#pragma unroll
  for (int i = 0; i < 8; ++i) {
    int q = t + i * 256, row = q >> 5, col = (q & 31) << 2;
    wOff[i] = col * WLD + row;
    (void)wRow;
  }

  // Hoisted per-lane fragment base pointers
  const uint4 *xA[2], *aA[2], *bgB[2], *buB[2], *bdB[2];
#pragma unroll
  for (int mi = 0; mi < 2; ++mi) {
    int m = mt0 + mi * 16 + lane16;
    xA[mi] = (const uint4*)xS   + m * (XLD / 8);
    aA[mi] = (const uint4*)actS + m * (ALD / 8);
  }
#pragma unroll
  for (int ni = 0; ni < 2; ++ni) {
    int n = nb0 + ni * 16 + lane16;
    bgB[ni] = (const uint4*)wgT + n * (WLD / 8);
    buB[ni] = (const uint4*)wuT + n * (WLD / 8);
    bdB[ni] = (const uint4*)wdT + n * (WLD / 8);
  }

  const v8f vzero = {0.f, 0.f, 0.f, 0.f, 0.f, 0.f, 0.f, 0.f};

  // ---------------- Phase 1: act = silu(x@Wg) * (x@Wu)  -> LDS (bf16) -------
  for (int nb = 0; nb < EDIM; nb += 128) {
    v8f accg[4], accu[4];                            // [mi*2+ni]
#pragma unroll
    for (int i = 0; i < 4; ++i) { accg[i] = vzero; accu[i] = vzero; }

    // uniform (scalar) tile bases; wrap on last iteration (redundant fetch)
    const float* xb = x + (size_t)m0 * HDIM;
    const float* gb = gW + nb;
    const float* ub = uW + nb;

    Raw1 raw;  Pk1 pk;
#pragma unroll
    for (int i = 0; i < 4; ++i) raw.xv[i] = ldg4(xb, offX[i]);
#pragma unroll
    for (int i = 0; i < 8; ++i) { raw.gv[i] = ldg4(gb, offW[i]);
                                  raw.uv[i] = ldg4(ub, offW[i]); }
    cvt1(pk, raw);

    for (int kb = 0; kb < HDIM; kb += 64) {
      __syncthreads();                               // LDS free
      // ---- pure ds_store section ----
#pragma unroll
      for (int i = 0; i < 4; ++i) *(uint2*)xSt[i] = pk.xv[i];
#pragma unroll
      for (int i = 0; i < 8; ++i) {
        unsigned short* g0 = wgT + wOff[i];
        unsigned short* u0 = wuT + wOff[i];
        st2(g0,           g0 + WLD,     pk.gv[i].x);
        st2(g0 + 2 * WLD, g0 + 3 * WLD, pk.gv[i].y);
        st2(u0,           u0 + WLD,     pk.uv[i].x);
        st2(u0 + 2 * WLD, u0 + 3 * WLD, pk.uv[i].y);
      }
      __syncthreads();                               // LDS ready

      // advance uniform bases (wrap keeps reads in-bounds on last iter)
      const bool last = (kb + 64 >= HDIM);
      xb = last ? x + (size_t)m0 * HDIM : xb + 64;
      gb = last ? gW + nb : gb + (size_t)64 * EDIM;
      ub = last ? uW + nb : ub + (size_t)64 * EDIM;
#pragma unroll
      for (int i = 0; i < 4; ++i) raw.xv[i] = ldg4(xb, offX[i]);
#pragma unroll
      for (int i = 0; i < 8; ++i) { raw.gv[i] = ldg4(gb, offW[i]);
                                    raw.uv[i] = ldg4(ub, offW[i]); }

#pragma unroll
      for (int kc = 0; kc < 64; kc += 32) {
        const int qa = (kc >> 3) + 2 * half;         // A frag uint4 index
        const int qb = (kc >> 3) + 2 * half;         // B frag uint4 index
        v16bf a0 = frag_a(xA[0], qa);
        v16bf a1 = frag_a(xA[1], qa);
        v16bf g0 = frag_b(bgB[0], qb);
        v16bf g1 = frag_b(bgB[1], qb);
        accg[0] = wmma_bf16(a0, g0, accg[0]);
        accg[2] = wmma_bf16(a1, g0, accg[2]);
        accg[1] = wmma_bf16(a0, g1, accg[1]);
        accg[3] = wmma_bf16(a1, g1, accg[3]);
        v16bf u0 = frag_b(buB[0], qb);
        v16bf u1 = frag_b(buB[1], qb);
        accu[0] = wmma_bf16(a0, u0, accu[0]);
        accu[2] = wmma_bf16(a1, u0, accu[2]);
        accu[1] = wmma_bf16(a0, u1, accu[1]);
        accu[3] = wmma_bf16(a1, u1, accu[3]);
      }
      cvt1(pk, raw);                                 // late wait on global loads
    }
    // silu(g)*u, f32 -> bf16, into resident act tile
#pragma unroll
    for (int mi = 0; mi < 2; ++mi)
#pragma unroll
      for (int ni = 0; ni < 2; ++ni)
#pragma unroll
        for (int i = 0; i < 8; ++i) {
          float g = accg[mi * 2 + ni][i];
          float u = accu[mi * 2 + ni][i];
          float s = g / (1.0f + __expf(-g));         // silu
          int mm = mt0 + mi * 16 + i + 8 * half;
          int nn = nb + nb0 + ni * 16 + lane16;
          actS[mm * ALD + nn] = f2bf(s * u);
        }
  }

  // ---------------- Phase 2: out = act @ Wd --------------------------------
  for (int hb = 0; hb < HDIM; hb += 128) {
    v8f acc[4];
#pragma unroll
    for (int i = 0; i < 4; ++i) acc[i] = vzero;

    const float* db = dW + hb;
    Raw2 raw;  Pk2 pk;
#pragma unroll
    for (int i = 0; i < 8; ++i) raw.dv[i] = ldg4(db, offD[i]);
    cvt2(pk, raw);

    for (int eb = 0; eb < EDIM; eb += 64) {
      __syncthreads();                               // also fences act stores
#pragma unroll
      for (int i = 0; i < 8; ++i) {
        unsigned short* d0 = wdT + wOff[i];
        st2(d0,           d0 + WLD,     pk.dv[i].x);
        st2(d0 + 2 * WLD, d0 + 3 * WLD, pk.dv[i].y);
      }
      __syncthreads();

      const bool last = (eb + 64 >= EDIM);
      db = last ? dW + hb : db + (size_t)64 * HDIM;
#pragma unroll
      for (int i = 0; i < 8; ++i) raw.dv[i] = ldg4(db, offD[i]);

#pragma unroll
      for (int kc = 0; kc < 64; kc += 32) {
        const int qa = ((eb + kc) >> 3) + 2 * half;
        const int qb = (kc >> 3) + 2 * half;
        v16bf a0 = frag_a(aA[0], qa);
        v16bf a1 = frag_a(aA[1], qa);
        v16bf b0 = frag_b(bdB[0], qb);
        v16bf b1 = frag_b(bdB[1], qb);
        acc[0] = wmma_bf16(a0, b0, acc[0]);
        acc[2] = wmma_bf16(a1, b0, acc[2]);
        acc[1] = wmma_bf16(a0, b1, acc[1]);
        acc[3] = wmma_bf16(a1, b1, acc[3]);
      }
      cvt2(pk, raw);
    }
#pragma unroll
    for (int mi = 0; mi < 2; ++mi)
#pragma unroll
      for (int ni = 0; ni < 2; ++ni)
#pragma unroll
        for (int i = 0; i < 8; ++i) {
          int mm = m0 + mt0 + mi * 16 + i + 8 * half;
          int hh = hb + nb0 + ni * 16 + lane16;
          out[((size_t)mm * KSEL + e) * HDIM + hh] = acc[mi * 2 + ni][i];
        }
  }
}

extern "C" void kernel_launch(void* const* d_in, const int* in_sizes, int n_in,
                              void* d_out, int out_size, void* d_ws, size_t ws_size,
                              hipStream_t stream) {
  (void)in_sizes; (void)n_in; (void)out_size; (void)d_ws; (void)ws_size;
  const float* x    = (const float*)d_in[0];
  const float* gate = (const float*)d_in[1];
  const float* up   = (const float*)d_in[2];
  const float* down = (const float*)d_in[3];
  const int*   eidx = (const int*)d_in[4];
  float*       out  = (float*)d_out;

  const size_t smem = (size_t)(MT * ALD + MT * XLD + 2 * 128 * WLD) *
                      sizeof(unsigned short);          // 284,672 B < 320 KB/WGP
  hipFuncSetAttribute((const void*)moe_swiglu_fused,
                      hipFuncAttributeMaxDynamicSharedMemorySize, (int)smem);

  dim3 grid(MROWS / MT, KSEL);
  moe_swiglu_fused<<<grid, dim3(256), smem, stream>>>(x, gate, up, down, eidx, out);
}
